// HybridMambaConvRefiner_84722524881480
// MI455X (gfx1250) — matrix operand order repaired and checked by reading, not compiled, not run on hardware
//
#include <hip/hip_runtime.h>
#include <hip/hip_bf16.h>

typedef __attribute__((ext_vector_type(16))) _Float16 v16h;
typedef __attribute__((ext_vector_type(8)))  _Float16 v8h;
typedef __attribute__((ext_vector_type(8)))  float    v8f;

#define CBATCH 8
#define CT     1000
#define CH     512
#define CKH    8      /* codebooks / heads */
#define CV     1024
#define CTD    512
#define CM     8000   /* CBATCH*CT rows */

// ---------------------------------------------------------------- small ops

__global__ void textcond_kernel(const float* __restrict__ te, const float* __restrict__ w,
                                const float* __restrict__ b, float* __restrict__ tc) {
  int bb = blockIdx.x, h = threadIdx.x;   // blockDim 512
  float s = b[h];
  for (int i = 0; i < CTD; ++i) s += te[bb * CTD + i] * w[i * CH + h];
  tc[bb * CH + h] = s;
}

__global__ void embed_kernel(const int* __restrict__ tok, const int* __restrict__ clvl,
                             const float* __restrict__ temb, const float* __restrict__ pos,
                             const float* __restrict__ cemb, const float* __restrict__ tc,
                             float* __restrict__ x) {
  int m = blockIdx.x, b = m / CT, t = m % CT;
  int cl = clvl[0]; if (cl > 19) cl = 19;
  for (int h = threadIdx.x; h < CH; h += blockDim.x) {
    float s = 0.f;
#pragma unroll
    for (int k = 0; k < CKH; ++k) {
      int tk = tok[(b * CKH + k) * CT + t];
      s += temb[((size_t)k * CV + tk) * CH + h];
    }
    s *= (1.0f / CKH);
    s += pos[t * CH + h] + cemb[cl * CH + h] + tc[b * CH + h];
    x[(size_t)m * CH + h] = s;
  }
}

__global__ __launch_bounds__(256) void rowmean_kernel(const float* __restrict__ x,
                                                      float* __restrict__ u) {
  __shared__ float red[256];
  int m = blockIdx.x, tid = threadIdx.x;
  red[tid] = x[(size_t)m * CH + tid] + x[(size_t)m * CH + tid + 256];
  __syncthreads();
  for (int o = 128; o > 0; o >>= 1) { if (tid < o) red[tid] += red[tid + o]; __syncthreads(); }
  if (tid == 0) u[m] = red[0] * (1.0f / CH);
}

// ------------------------------------------------- SSM as causal convolution

__global__ __launch_bounds__(512) void ssm_kernelgen(const float* __restrict__ A,
                                                     const float* __restrict__ Bv,
                                                     const float* __restrict__ Cv,
                                                     float* __restrict__ karr) {
  __shared__ float w[CH];
  __shared__ float red[CH];
  int i = blockIdx.x, j = threadIdx.x;
  const float* Ai = A + (size_t)i * CH * CH;
  float Cj = Cv[i * CH + j];
  w[j] = Bv[i * CH + j];
  __syncthreads();
  for (int tau = 0; tau < CT; ++tau) {
    red[j] = Cj * w[j]; __syncthreads();
    for (int o = 256; o > 0; o >>= 1) { if (j < o) red[j] += red[j + o]; __syncthreads(); }
    if (j == 0) karr[i * CT + tau] = red[0];
    const float* Aj = Ai + (size_t)j * CH;
    float nw = 0.f;
    for (int k = 0; k < CH; ++k) nw += Aj[k] * w[k];
    __syncthreads();
    w[j] = nw;
    __syncthreads();
  }
}

// u[] at LDS offset 0, kk[] at LDS offset CT*4 (declaration order).
__global__ __launch_bounds__(1024) void ssm_apply(const float* __restrict__ u0,
                                                  const float* __restrict__ karr,
                                                  const float* __restrict__ Dv,
                                                  float* __restrict__ delta) {
  __shared__ float u[CT];
  __shared__ float kk[CT];
  int b = blockIdx.x, t = threadIdx.x;
  float dl = 0.f;
  if (t < CT) {
    unsigned uoff = (unsigned)(t * 4);
    const float* gp = u0 + b * CT + t;
    // async copy global -> LDS (ASYNCcnt-tracked, no VGPR round-trip)
    asm volatile("global_load_async_to_lds_b32 %0, %1, off"
                 :: "v"(uoff), "v"(gp) : "memory");
  }
  asm volatile("s_wait_asynccnt 0" ::: "memory");
  __syncthreads();
  for (int i = 0; i < 3; ++i) {
    if (t < CT) {
      unsigned koff = (unsigned)(CT * 4 + t * 4);
      const float* gk = karr + i * CT + t;
      asm volatile("global_load_async_to_lds_b32 %0, %1, off"
                   :: "v"(koff), "v"(gk) : "memory");
    }
    asm volatile("s_wait_asynccnt 0" ::: "memory");
    __syncthreads();
    float y = 0.f;
    if (t < CT) for (int s = 0; s <= t; ++s) y += kk[t - s] * u[s];
    __syncthreads();
    if (t < CT) { float add = y + u[t] * Dv[i]; dl += add; u[t] += add; }
    __syncthreads();
  }
  if (t < CT) delta[b * CT + t] = dl;
}

__global__ void ff_write(const float* __restrict__ x, const float* __restrict__ delta,
                         float* __restrict__ ffo, _Float16* __restrict__ fused) {
  int m = blockIdx.x;
  float d = delta[m];
  for (int h = threadIdx.x; h < CH; h += blockDim.x) {
    float v = x[(size_t)m * CH + h] + d;
    ffo[(size_t)m * CH + h] = v;
    fused[(size_t)m * (2 * CH) + h] = (_Float16)v;
  }
}

// ------------------------------------------------- dilated grouped conv

__global__ __launch_bounds__(512) void conv_dilated(const float* __restrict__ x,
                                                    const float* __restrict__ w,
                                                    const float* __restrict__ cb,
                                                    _Float16* __restrict__ msT) {
  int m = blockIdx.x, b = m / CT, t = m % CT;
  int c = threadIdx.x, j = c >> 7, o = c & 127;
  int d = (j == 0) ? 1 : (j == 1) ? 3 : (j == 2) ? 9 : 27;
  const float* wp = w + ((size_t)(j * 128 + o) * 128) * 3;
  float acc = cb[j * 128 + o];
#pragma unroll
  for (int kk = 0; kk < 3; ++kk) {
    int tt = t + (kk - 1) * d;
    if (tt < 0 || tt >= CT) continue;
    const float* xr = x + (size_t)(b * CT + tt) * CH + j * 128;
    for (int i = 0; i < 128; ++i) acc += wp[i * 3 + kk] * xr[i];
  }
  msT[(size_t)m * CH + c] = (_Float16)acc;
}

// ------------------------------------------------- WMMA GEMM (f16 -> f32)
// A: [M,K] f16 row-major.  Bt: [N,K] f16 (weights pre-transposed so each
// lane's B fragment is 16 contiguous halves).  Block = 4 waves -> 64x64 tile;
// each wave computes a 16x64 strip (4 WMMA tiles sharing one A fragment).

__global__ __launch_bounds__(128)
void gemm_wmma_kernel(const _Float16* __restrict__ A, const _Float16* __restrict__ Bt,
                      const float* __restrict__ bias, const float* __restrict__ resid,
                      float* __restrict__ out32, _Float16* __restrict__ out16,
                      int M, int N, int K, int act_silu, int row_seg, int row_skip) {
  const int wave = threadIdx.x >> 5;
  const int lane = threadIdx.x & 31;
  const int m0 = (blockIdx.y << 6) + (wave << 4);
  const int n0 = blockIdx.x << 6;

  const int mA  = lane & 15;
  const int kbA = (lane >> 4) << 4;    // A lane K base: 0 or 16 (contiguous)
  const int nL  = lane & 15;
  const int kbB = (lane >> 4) << 4;    // B lane K base: 0 or 16

  const _Float16* Arow = A + (size_t)(m0 + mA) * K + kbA;
  const _Float16* Br0 = Bt + (size_t)(n0 + nL) * K + kbB;
  const _Float16* Br1 = Br0 + (size_t)16 * K;
  const _Float16* Br2 = Br0 + (size_t)32 * K;
  const _Float16* Br3 = Br0 + (size_t)48 * K;

  v8f acc0 = {}, acc1 = {}, acc2 = {}, acc3 = {};
  for (int k0 = 0; k0 < K; k0 += 32) {
    __builtin_prefetch(Arow + k0 + 128, 0, 3);    // global_prefetch_b8 (near)
    __builtin_prefetch(Br0 + k0 + 128, 0, 3);
    v8h alo = *(const v8h*)(Arow + k0);
    v8h ahi = *(const v8h*)(Arow + k0 + 8);
    v16h a;
#pragma unroll
    for (int e = 0; e < 8; ++e) { a[e] = alo[e]; a[e + 8] = ahi[e]; }
    v16h b0 = *(const v16h*)(Br0 + k0);
    v16h b1 = *(const v16h*)(Br1 + k0);
    v16h b2 = *(const v16h*)(Br2 + k0);
    v16h b3 = *(const v16h*)(Br3 + k0);
    acc0 = __builtin_amdgcn_wmma_f32_16x16x32_f16(false, a, false, b0, (short)0, acc0, false, false);
    acc1 = __builtin_amdgcn_wmma_f32_16x16x32_f16(false, a, false, b1, (short)0, acc1, false, false);
    acc2 = __builtin_amdgcn_wmma_f32_16x16x32_f16(false, a, false, b2, (short)0, acc2, false, false);
    acc3 = __builtin_amdgcn_wmma_f32_16x16x32_f16(false, a, false, b3, (short)0, acc3, false, false);
  }

  const int rbase = m0 + ((lane >> 4) << 3);
  v8f accs[4] = {acc0, acc1, acc2, acc3};
#pragma unroll
  for (int nt = 0; nt < 4; ++nt) {
    const int col = n0 + nt * 16 + nL;
#pragma unroll
    for (int r = 0; r < 8; ++r) {
      int row = rbase + r;
      float v = accs[nt][r];
      if (bias)  v += bias[col];
      if (resid) v += resid[(size_t)row * N + col];
      if (act_silu) v = v / (1.0f + __expf(-v));
      if (out32) {
        size_t orow = row_skip ? (size_t)row + (size_t)(row / row_seg) * row_skip : (size_t)row;
        out32[orow * (size_t)N + col] = v;
      }
      if (out16) out16[(size_t)row * N + col] = (_Float16)v;
    }
  }
}

// ------------------------------------------------- LayerNorm

__global__ __launch_bounds__(256) void ln_kernel(const float* __restrict__ bw,
                                                 const float* __restrict__ lw,
                                                 const float* __restrict__ lb,
                                                 float* __restrict__ bwt,
                                                 _Float16* __restrict__ fused) {
  __shared__ float red[256];
  __shared__ float sMu, sRs;
  int m = blockIdx.x, tid = threadIdx.x;
  const float* rp = bw + (size_t)m * CH;
  float a = rp[tid], b = rp[tid + 256];
  red[tid] = a + b; __syncthreads();
  for (int o = 128; o > 0; o >>= 1) { if (tid < o) red[tid] += red[tid + o]; __syncthreads(); }
  if (tid == 0) sMu = red[0] * (1.0f / CH);
  __syncthreads();
  float mu = sMu;
  red[tid] = (a - mu) * (a - mu) + (b - mu) * (b - mu); __syncthreads();
  for (int o = 128; o > 0; o >>= 1) { if (tid < o) red[tid] += red[tid + o]; __syncthreads(); }
  if (tid == 0) sRs = rsqrtf(red[0] * (1.0f / CH) + 1e-5f);
  __syncthreads();
  float rs = sRs;
  float v0 = (a - mu) * rs * lw[tid] + lb[tid];
  float v1 = (b - mu) * rs * lw[tid + 256] + lb[tid + 256];
  bwt[(size_t)m * CH + tid] = v0;
  bwt[(size_t)m * CH + tid + 256] = v1;
  fused[(size_t)m * (2 * CH) + CH + tid] = (_Float16)v0;
  fused[(size_t)m * (2 * CH) + CH + tid + 256] = (_Float16)v1;
}

// ------------------------------------------------- confidence head tail

__global__ __launch_bounds__(256) void conf_kernel(const _Float16* __restrict__ c1,
                                                   const float* __restrict__ w2,
                                                   const float* __restrict__ b2,
                                                   float* __restrict__ conf) {
  int warp = threadIdx.x >> 5, lane = threadIdx.x & 31;
  int m = blockIdx.x * 8 + warp;
  if (m >= CM) return;
  float s = 0.f;
#pragma unroll
  for (int i = 0; i < 8; ++i) {
    int c = lane * 8 + i;
    s += (float)c1[(size_t)m * 256 + c] * w2[c];
  }
  for (int o = 16; o; o >>= 1) s += __shfl_xor(s, o, 32);
  if (lane == 0) conf[m] = 1.0f / (1.0f + __expf(-(s + b2[0])));
}

// ------------------------------------------------- fl / bl reductions

__global__ __launch_bounds__(256) void fl_partial(const float* __restrict__ ff,
                                                  float* __restrict__ part) {
  __shared__ float red[256];
  const long NTOT = (long)CBATCH * (CT - 1) * CH;
  float s = 0.f;
  for (long e = (long)blockIdx.x * 256 + threadIdx.x; e < NTOT; e += (long)gridDim.x * 256) {
    long b = e / ((long)(CT - 1) * CH);
    long r = e % ((long)(CT - 1) * CH);
    long t = r / CH, h = r % CH;
    float d = ff[((b * CT) + t + 1) * CH + h] - ff[((b * CT) + t) * CH + h];
    s += d * d;
  }
  red[threadIdx.x] = s; __syncthreads();
  for (int o = 128; o > 0; o >>= 1) { if (threadIdx.x < o) red[threadIdx.x] += red[threadIdx.x + o]; __syncthreads(); }
  if (threadIdx.x == 0) part[blockIdx.x] = red[0];
}

__global__ __launch_bounds__(512) void fl_final(const float* __restrict__ part,
                                                float* __restrict__ out) {
  __shared__ float red[512];
  red[threadIdx.x] = part[threadIdx.x]; __syncthreads();
  for (int o = 256; o > 0; o >>= 1) { if (threadIdx.x < o) red[threadIdx.x] += red[threadIdx.x + o]; __syncthreads(); }
  if (threadIdx.x == 0)
    out[0] = red[0] / ((float)CBATCH * (CT - 1) * CH) * 0.001f;
}

__global__ __launch_bounds__(256) void bl_stage1(const float* __restrict__ bwt,
                                                 float* __restrict__ sums) {
  __shared__ float rs[256], rq[256];
  int bh = blockIdx.x, b = bh >> 9, h = bh & 511;
  float s = 0.f, q = 0.f;
  for (int t = threadIdx.x; t < CT; t += 256) {
    float v = bwt[((size_t)(b * CT) + t) * CH + h];
    s += v; q += v * v;
  }
  rs[threadIdx.x] = s; rq[threadIdx.x] = q; __syncthreads();
  for (int o = 128; o > 0; o >>= 1) {
    if (threadIdx.x < o) { rs[threadIdx.x] += rs[threadIdx.x + o]; rq[threadIdx.x] += rq[threadIdx.x + o]; }
    __syncthreads();
  }
  if (threadIdx.x == 0) { sums[bh * 2] = rs[0]; sums[bh * 2 + 1] = rq[0]; }
}

__global__ __launch_bounds__(1024) void bl_final(const float* __restrict__ sums,
                                                 float* __restrict__ out) {
  __shared__ float rq[1024], rv[1024];
  float q = 0.f, v = 0.f;
  for (int i = threadIdx.x; i < 4096; i += 1024) {
    float s = sums[i * 2], ss = sums[i * 2 + 1];
    q += ss;
    v += (ss - s * s * (1.0f / CT)) * (1.0f / (CT - 1));
  }
  rq[threadIdx.x] = q; rv[threadIdx.x] = v; __syncthreads();
  for (int o = 512; o > 0; o >>= 1) {
    if (threadIdx.x < o) { rq[threadIdx.x] += rq[threadIdx.x + o]; rv[threadIdx.x] += rv[threadIdx.x + o]; }
    __syncthreads();
  }
  if (threadIdx.x == 0)
    out[0] = 1e-4f * (rq[0] / ((float)CBATCH * CT * CH)) - 1e-4f * (rv[0] / 4096.0f);
}

// ------------------------------------------------- conversions

__global__ void cvt_f32_f16(const float* __restrict__ in, _Float16* __restrict__ o, long n) {
  long i = (long)blockIdx.x * blockDim.x + threadIdx.x;
  if (i < n) o[i] = (_Float16)in[i];
}

// in: [R][C] f32 row-major  ->  out: [C][R] f16 (transposed)
__global__ void transpose_cvt(const float* __restrict__ in, _Float16* __restrict__ o,
                              int R, int C) {
  long idx = (long)blockIdx.x * blockDim.x + threadIdx.x;
  if (idx >= (long)R * C) return;
  int c = (int)(idx / R), r = (int)(idx % R);
  o[idx] = (_Float16)in[(long)r * C + c];
}

// ---------------------------------------------------------------- launcher

extern "C" void kernel_launch(void* const* d_in, const int* in_sizes, int n_in,
                              void* d_out, int out_size, void* d_ws, size_t ws_size,
                              hipStream_t stream) {
  const float* text_emb = (const float*)d_in[0];
  const int*   tokens   = (const int*)d_in[1];
  const int*   clvl     = (const int*)d_in[2];
  const float* token_emb= (const float*)d_in[3];
  const float* pos_emb  = (const float*)d_in[4];
  const float* conf_emb = (const float*)d_in[5];
  const float* text_w   = (const float*)d_in[6];
  const float* text_b   = (const float*)d_in[7];
  const float* ssm_A    = (const float*)d_in[8];
  const float* ssm_B    = (const float*)d_in[9];
  const float* ssm_C    = (const float*)d_in[10];
  const float* ssm_D    = (const float*)d_in[11];
  const float* conv_w   = (const float*)d_in[12];
  const float* conv_b   = (const float*)d_in[13];
  const float* pw_w     = (const float*)d_in[14];
  const float* pw_b     = (const float*)d_in[15];
  const float* ln_w     = (const float*)d_in[16];
  const float* ln_b     = (const float*)d_in[17];
  const float* fus_w1   = (const float*)d_in[18];
  const float* fus_b1   = (const float*)d_in[19];
  const float* fus_w2   = (const float*)d_in[20];
  const float* fus_b2   = (const float*)d_in[21];
  const float* head_w1  = (const float*)d_in[22];
  const float* head_b1  = (const float*)d_in[23];
  const float* head_w2  = (const float*)d_in[24];
  const float* head_b2  = (const float*)d_in[25];
  const float* cf_w1    = (const float*)d_in[26];
  const float* cf_b1    = (const float*)d_in[27];
  const float* cf_w2    = (const float*)d_in[28];
  const float* cf_b2    = (const float*)d_in[29];

  float* out        = (float*)d_out;
  float* out_logits = out;                         // B*K*T*V = 65,536,000
  float* out_conf   = out + 65536000;              // 8000
  float* out_fl     = out + 65544000;              // 1
  float* out_bl     = out + 65544001;              // 1
  float* out_ff     = out + 65544002;              // 4,096,000
  float* out_bwt    = out + 69640002;              // 4,096,000

  char* p = (char*)d_ws;
  auto alloc = [&](size_t bytes) -> void* {
    void* r = (void*)p; p += (bytes + 255) & ~(size_t)255; return r;
  };
  float*    tc     = (float*)alloc((size_t)CBATCH * CH * 4);
  float*    xbuf   = (float*)alloc((size_t)CM * CH * 4);
  float*    u0     = (float*)alloc((size_t)CM * 4);
  float*    karr   = (float*)alloc((size_t)3 * CT * 4);
  float*    delta  = (float*)alloc((size_t)CM * 4);
  float*    bw32   = (float*)alloc((size_t)CM * CH * 4);
  float*    fpart  = (float*)alloc((size_t)512 * 4);
  float*    blsum  = (float*)alloc((size_t)4096 * 2 * 4);
  _Float16* msT    = (_Float16*)alloc((size_t)CM * CH * 2);
  _Float16* fusedA = (_Float16*)alloc((size_t)CM * 2 * CH * 2);
  _Float16* s_h    = (_Float16*)alloc((size_t)CM * CH * 2);
  _Float16* fin_h  = (_Float16*)alloc((size_t)CM * CH * 2);
  _Float16* hid_h  = (_Float16*)alloc((size_t)CM * CH * 2);
  _Float16* c1_h   = (_Float16*)alloc((size_t)CM * (CH / 2) * 2);
  // weights, all stored TRANSPOSED as [N][K] f16
  _Float16* pw_wT  = (_Float16*)alloc((size_t)CH * CH * 2);            // already [n][k]
  _Float16* fw1t   = (_Float16*)alloc((size_t)CH * 2 * CH * 2);        // [512][1024]
  _Float16* fw2t   = (_Float16*)alloc((size_t)CH * CH * 2);            // [512][512]
  _Float16* hw1t   = (_Float16*)alloc((size_t)CKH * CH * CH * 2);      // [k][512][512]
  _Float16* hw2t   = (_Float16*)alloc((size_t)CKH * CV * CH * 2);      // [k][1024][512]
  _Float16* cw1t   = (_Float16*)alloc((size_t)(CH / 2) * CH * 2);      // [256][512]

  auto cvt = [&](const float* src, _Float16* dst, long n) {
    cvt_f32_f16<<<dim3((unsigned)((n + 255) / 256)), dim3(256), 0, stream>>>(src, dst, n);
  };
  auto tcv = [&](const float* src, _Float16* dst, int R, int C) {
    long n = (long)R * C;
    transpose_cvt<<<dim3((unsigned)((n + 255) / 256)), dim3(256), 0, stream>>>(src, dst, R, C);
  };
  // weight conversions (pw_w[c][c'] is already the [n][k] layout we need)
  cvt(pw_w, pw_wT, (long)CH * CH);
  tcv(fus_w1, fw1t, 2 * CH, CH);
  tcv(fus_w2, fw2t, CH, CH);
  for (int k = 0; k < CKH; ++k) {
    tcv(head_w1 + (size_t)k * CH * CH, hw1t + (size_t)k * CH * CH, CH, CH);
    tcv(head_w2 + (size_t)k * CH * CV, hw2t + (size_t)k * CV * CH, CH, CV);
  }
  tcv(cf_w1, cw1t, CH, CH / 2);

  // embedding path
  textcond_kernel<<<dim3(CBATCH), dim3(CH), 0, stream>>>(text_emb, text_w, text_b, tc);
  embed_kernel<<<dim3(CM), dim3(256), 0, stream>>>(tokens, clvl, token_emb, pos_emb,
                                                   conf_emb, tc, xbuf);
  // SSM branch (scalar-channel reformulation: k_tau = C A^tau B causal conv)
  rowmean_kernel<<<dim3(CM), dim3(256), 0, stream>>>(xbuf, u0);
  ssm_kernelgen<<<dim3(3), dim3(512), 0, stream>>>(ssm_A, ssm_B, ssm_C, karr);
  ssm_apply<<<dim3(CBATCH), dim3(1024), 0, stream>>>(u0, karr, ssm_D, delta);
  ff_write<<<dim3(CM), dim3(256), 0, stream>>>(xbuf, delta, out_ff, fusedA);

  auto gemm = [&](const _Float16* Am, const _Float16* Bm, const float* bias,
                  const float* resid, float* o32, _Float16* o16,
                  int N, int K, int silu, int rseg, int rskip) {
    dim3 g((unsigned)(N / 64), (unsigned)(CM / 64));
    gemm_wmma_kernel<<<g, dim3(128), 0, stream>>>(Am, Bm, bias, resid, o32, o16,
                                                  CM, N, K, silu, rseg, rskip);
  };

  // conv branch
  conv_dilated<<<dim3(CM), dim3(512), 0, stream>>>(xbuf, conv_w, conv_b, msT);
  gemm(msT, pw_wT, pw_b, /*resid=*/xbuf, bw32, nullptr, CH, CH, 0, 1, 0);
  ln_kernel<<<dim3(CM), dim3(256), 0, stream>>>(bw32, ln_w, ln_b, out_bwt, fusedA);

  // fusion MLP
  gemm(fusedA, fw1t, fus_b1, nullptr, nullptr, s_h, CH, 2 * CH, 1, 1, 0);
  gemm(s_h, fw2t, fus_b2, nullptr, nullptr, fin_h, CH, CH, 0, 1, 0);

  // per-head GEMMs -> logits (strided row map: (b*K+k)*T + t)
  for (int k = 0; k < CKH; ++k) {
    gemm(fin_h, hw1t + (size_t)k * CH * CH, head_b1 + (size_t)k * CH,
         nullptr, nullptr, hid_h, CH, CH, 1, 1, 0);
    gemm(hid_h, hw2t + (size_t)k * CV * CH, head_b2 + (size_t)k * CV,
         nullptr, out_logits + (size_t)k * CT * CV, nullptr,
         CV, CH, 0, /*row_seg=*/CT, /*row_skip=*/(CKH - 1) * CT);
  }

  // confidence head
  gemm(fin_h, cw1t, cf_b1, nullptr, nullptr, c1_h, CH / 2, CH, 1, 1, 0);
  conf_kernel<<<dim3(CM / 8), dim3(256), 0, stream>>>(c1_h, cf_w2, cf_b2, out_conf);

  // auxiliary losses (deterministic two-stage reductions)
  fl_partial<<<dim3(512), dim3(256), 0, stream>>>(out_ff, fpart);
  fl_final<<<dim3(1), dim3(512), 0, stream>>>(fpart, out_fl);
  bl_stage1<<<dim3(4096), dim3(256), 0, stream>>>(out_bwt, blsum);
  bl_final<<<dim3(1), dim3(1024), 0, stream>>>(blsum, out_bl);
}
